// EfficientMultiScaleAttention_79809082294688
// MI455X (gfx1250) — matrix operand order repaired and checked by reading, not compiled, not run on hardware
//
#include <hip/hip_runtime.h>
#include <math.h>

// Problem constants (from reference setup_inputs)
#define BG   256   // B*g
#define HH   64
#define WW   48
#define DD   32    // channels per group
#define CC   256
#define ROWS 4     // output rows per k_main block
#define HB   (HH / ROWS)   // 16 row-blocks
#define LN_EPSF 1e-3f

typedef __attribute__((ext_vector_type(16))) _Float16 v16h;
typedef __attribute__((ext_vector_type(4)))  _Float16 v4h;
typedef __attribute__((ext_vector_type(8)))  float    v8f;

// ---------------- workspace layout (floats) ----------------
#define OFF_HWP  ((size_t)0)                       // [BG][112][32] pooled pre-act
#define OFF_HWG  (OFF_HWP + (size_t)BG*112*DD)     // [BG][112][32] sigmoid gates
#define OFF_X11  (OFF_HWG + (size_t)BG*112*DD)     // [BG][32]
#define OFF_X21  (OFF_X11 + (size_t)BG*DD)         // [BG][32]
#define OFF_PS1  (OFF_X21 + (size_t)BG*DD)         // [BG][HB][32] partial sums of x1
#define OFF_PS2  (OFF_PS1 + (size_t)BG*HB*DD)      // [BG][HB][32] partial sums of x2
#define OFF_WF3  (OFF_PS2 + (size_t)BG*HB*DD)      // 9216 halfs (4608 f): conv3x3 B-frags
#define OFF_WF1  (OFF_WF3 + (size_t)4608)          // 1024 halfs (512 f): conv1x1 B-frags
#define OFF_X2   (OFF_WF1 + (size_t)512)           // [BG][HH][WW][32]

__device__ __forceinline__ float wave_sum32(float v) {
#pragma unroll
  for (int m = 16; m > 0; m >>= 1) v += __shfl_xor(v, m, 32);
  return v;
}
__device__ __forceinline__ float wave_max32(float v) {
#pragma unroll
  for (int m = 16; m > 0; m >>= 1) v = fmaxf(v, __shfl_xor(v, m, 32));
  return v;
}
__device__ __forceinline__ float fast_rcp(float v) { return __builtin_amdgcn_rcpf(v); }
__device__ __forceinline__ float fast_rsq(float v) { return __builtin_amdgcn_rsqf(v); }
__device__ __forceinline__ float sigmoidf(float v) {
  return fast_rcp(1.0f + __expf(-v));
}

// ------------------------------------------------------------------
// K0: pre-swizzle both weight matrices into per-lane f16 WMMA
// B-fragment layout: frag[(tap)*2+nt][lane][e] = w[tap][K][N],
// K = 16*(lane>=16)+e (contiguous), N = nt*16 + (lane&15).
// ------------------------------------------------------------------
__global__ void k_wprep(const float* __restrict__ w3, const float* __restrict__ w1,
                        _Float16* __restrict__ wf3, _Float16* __restrict__ wf1) {
  int i = blockIdx.x * blockDim.x + threadIdx.x;
  if (i < 9 * 2 * 32 * 16) {
    int e = i & 15, lane = (i >> 4) & 31, nt = (i >> 9) & 1, tap = i >> 10;
    int k = ((lane & 16) ? 16 : 0) + e;
    int n = nt * 16 + (lane & 15);
    wf3[i] = (_Float16)w3[(tap * DD + k) * DD + n];
  }
  if (i < 2 * 32 * 16) {
    int e = i & 15, lane = (i >> 4) & 31, nt = i >> 9;
    int k = ((lane & 16) ? 16 : 0) + e;
    int n = nt * 16 + (lane & 15);
    wf1[i] = (_Float16)w1[k * DD + n];
  }
}

// ------------------------------------------------------------------
// K1: single-pass pooled means: x read exactly once.
// Thread t: channel c = t&31, w-base wb = t>>5 (w = wb+8k, k=0..5).
// Column (over-H) sums accumulate in registers; row (over-W) sums via
// one small LDS reduce per h.
// ------------------------------------------------------------------
__global__ void __launch_bounds__(256) k_pool(const float* __restrict__ x,
                                              float* __restrict__ hwp) {
  __shared__ float red[256];
  int bg = blockIdx.x;
  int b = bg >> 3, gi = bg & 7;
  const float* xb = x + (size_t)b * HH * WW * CC + (size_t)gi * DD;
  float* o = hwp + (size_t)bg * (HH + WW) * DD;
  int t = threadIdx.x;
  int c = t & 31, wb = t >> 5;
  float colacc[6] = {0.f, 0.f, 0.f, 0.f, 0.f, 0.f};
  for (int h = 0; h < HH; ++h) {
    float rp = 0.f;
#pragma unroll
    for (int k = 0; k < 6; ++k) {
      float v = xb[((size_t)h * WW + wb + 8 * k) * CC + c];
      colacc[k] += v;
      rp += v;
    }
    red[t] = rp;
    __syncthreads();
    if (t < 32) {
      float s = 0.f;
#pragma unroll
      for (int j = 0; j < 8; ++j) s += red[c + 32 * j];
      o[h * DD + c] = s * (1.0f / WW);
    }
    __syncthreads();
  }
#pragma unroll
  for (int k = 0; k < 6; ++k)
    o[(HH + wb + 8 * k) * DD + c] = colacc[k] * (1.0f / HH);
}

// ------------------------------------------------------------------
// K2: [BG*112, 32] @ [32,32] + bias -> sigmoid, via WMMA f16->f32.
// One wave per 16-row M-tile; B fragments pre-swizzled (v16h loads).
// ------------------------------------------------------------------
__global__ void __launch_bounds__(32) k_gate_gemm(
    const float* __restrict__ hwp, const _Float16* __restrict__ wf1,
    const float* __restrict__ b1, float* __restrict__ hwg) {
  int r0 = blockIdx.x * 16;
  int lane = threadIdx.x & 31;
  int m = lane & 15;
  int kbA = (lane & 16) ? 16 : 0;  // A frag: K = kbA + 0..15

  v16h a;
  const float* arow = hwp + (size_t)(r0 + m) * DD;
#pragma unroll
  for (int e = 0; e < 16; ++e) {
    int k = kbA + e;
    a[e] = (_Float16)arow[k];
  }
  const v16h* wfv = (const v16h*)wf1;
  v16h vb0 = wfv[0 * 32 + lane];
  v16h vb1 = wfv[1 * 32 + lane];

  v8f acc0 = {}, acc1 = {};
  acc0 = __builtin_amdgcn_wmma_f32_16x16x32_f16(false, a, false, vb0, (short)0, acc0, false, false);
  acc1 = __builtin_amdgcn_wmma_f32_16x16x32_f16(false, a, false, vb1, (short)0, acc1, false, false);

  int Mo = (lane & 16) ? 8 : 0;  // C/D: VGPR i -> M = i (+8 for upper lanes)
  int n = lane & 15;
  float bi0 = b1[n], bi1 = b1[16 + n];
#pragma unroll
  for (int i = 0; i < 8; ++i) {
    size_t r = (size_t)(r0 + Mo + i);
    hwg[r * DD + n]      = sigmoidf(acc0[i] + bi0);
    hwg[r * DD + 16 + n] = sigmoidf(acc1[i] + bi1);
  }
}

// ------------------------------------------------------------------
// K3: per (bg, 4-row band): 3x3 conv via implicit-GEMM WMMA -> x2,
// plus deterministic per-band partial sums of x1 (recomputed LN) and x2.
// Block = 96 threads (3 waves); wave w owns the 16-pixel M-tile w.
// Loop order: tap outer (B fragment loaded once), 4 row accumulators live.
// ------------------------------------------------------------------
__global__ void __launch_bounds__(96) k_main(
    const float* __restrict__ x, const _Float16* __restrict__ wfrag,
    const float* __restrict__ b3, const float* __restrict__ gam,
    const float* __restrict__ bet, const float* __restrict__ hwg,
    float* __restrict__ x2, float* __restrict__ ps1, float* __restrict__ ps2) {
  __shared__ __align__(32) _Float16 s_in[ROWS + 2][WW + 2][DD];  // padded rows (f16)
  __shared__ float red1[3][DD];
  __shared__ float red2[3][DD];

  int h0 = blockIdx.x * ROWS, bg = blockIdx.y;
  int b = bg >> 3, gi = bg & 7;
  const float* xb = x + (size_t)b * HH * WW * CC + (size_t)gi * DD;
  int tid = threadIdx.x;

  // stage ROWS+2 input rows with zero halo, float4 loads -> f16 LDS
  for (int i = tid; i < (ROWS + 2) * (WW + 2) * 8; i += 96) {
    int q = i & 7;                 // 4-channel granule within pixel
    int p = (i >> 3) % (WW + 2);
    int r = i / ((WW + 2) * 8);
    int hh = h0 - 1 + r, ww = p - 1;
    float4 v = make_float4(0.f, 0.f, 0.f, 0.f);
    if (hh >= 0 && hh < HH && ww >= 0 && ww < WW)
      v = *(const float4*)&xb[((size_t)hh * WW + ww) * CC + q * 4];
    v4h hv;
    hv[0] = (_Float16)v.x; hv[1] = (_Float16)v.y;
    hv[2] = (_Float16)v.z; hv[3] = (_Float16)v.w;
    *(v4h*)&s_in[r][p][q * 4] = hv;
  }
  __syncthreads();

  int wv = tid >> 5, lane = tid & 31;
  int m = lane & 15;
  int kbA = (lane & 16) ? 16 : 0;
  int Mo = (lane & 16) ? 8 : 0;
  int n = lane & 15;
  float bi0 = b3[n], bi1 = b3[16 + n];
  const v16h* wf = (const v16h*)wfrag;  // [(tap*2+nt)*32 + lane]

  // ---- conv 3x3: tap-major, K = 9 taps x 32ch, M-tile = 16 px, N = 32 ----
  v8f acc0[ROWS], acc1[ROWS];
#pragma unroll
  for (int hr = 0; hr < ROWS; ++hr) { acc0[hr] = (v8f){}; acc1[hr] = (v8f){}; }
#pragma unroll
  for (int tap = 0; tap < 9; ++tap) {
    int dy = tap / 3, dxo = tap % 3;
    v16h bb0 = wf[(tap * 2 + 0) * 32 + lane];
    v16h bb1 = wf[(tap * 2 + 1) * 32 + lane];
#pragma unroll
    for (int hr = 0; hr < ROWS; ++hr) {
      const _Float16* pa = &s_in[hr + dy][wv * 16 + m + dxo][kbA];
      v16h a;
      *(float4*)&a       = *(const float4*)pa;         // K = kbA + 0..7
      *((float4*)&a + 1) = *(const float4*)(pa + 8);   // K = kbA + 8..15
      acc0[hr] = __builtin_amdgcn_wmma_f32_16x16x32_f16(false, a, false, bb0, (short)0, acc0[hr], false, false);
      acc1[hr] = __builtin_amdgcn_wmma_f32_16x16x32_f16(false, a, false, bb1, (short)0, acc1[hr], false, false);
    }
  }

  // ---- write x2 (+bias), accumulate per-channel partial sums ----
  float p0acc = 0.f, p1acc = 0.f;
#pragma unroll
  for (int hr = 0; hr < ROWS; ++hr) {
    float* x2row = x2 + ((size_t)bg * HH + h0 + hr) * WW * DD;
#pragma unroll
    for (int i = 0; i < 8; ++i) {
      int w = wv * 16 + Mo + i;
      float v0 = acc0[hr][i] + bi0, v1 = acc1[hr][i] + bi1;
      x2row[(size_t)w * DD + n]      = v0;
      x2row[(size_t)w * DD + 16 + n] = v1;
      p0acc += v0;
      p1acc += v1;
    }
  }
  p0acc += __shfl_xor(p0acc, 16, 32);  // combine the two pixel halves per channel
  p1acc += __shfl_xor(p1acc, 16, 32);
  if (lane < 16) { red2[wv][n] = p0acc; red2[wv][16 + n] = p1acc; }

  // ---- x1 partial sums: recompute gated-LN per pixel (lane = channel) ----
  int c = lane;
  const float* hwb = hwg + (size_t)bg * (HH + WW) * DD;
  float gm = gam[c], bt = bet[c];
  float acc = 0.f;
  for (int hr = 0; hr < ROWS; ++hr) {
    float xhv = hwb[(h0 + hr) * DD + c];
    for (int p = wv; p < WW; p += 3) {
      float gxv = (float)s_in[hr + 1][p + 1][c];
      float y = gxv * xhv * hwb[(HH + p) * DD + c];
      float mu = wave_sum32(y) * (1.0f / DD);
      float e2 = wave_sum32(y * y) * (1.0f / DD);
      float x1v = (y - mu) * fast_rsq(e2 - mu * mu + LN_EPSF) * gm + bt;
      acc += x1v;
    }
  }
  red1[wv][c] = acc;
  __syncthreads();
  if (tid < 32) {
    size_t o = ((size_t)bg * HB + blockIdx.x) * DD + tid;
    ps1[o] = red1[0][tid] + red1[1][tid] + red1[2][tid];
    ps2[o] = red2[0][tid] + red2[1][tid] + red2[2][tid];
  }
}

// ------------------------------------------------------------------
// K4: reduce per-band partials (fixed order) and channel softmax
// ------------------------------------------------------------------
__global__ void __launch_bounds__(32) k_softmax(
    const float* __restrict__ ps1, const float* __restrict__ ps2,
    float* __restrict__ x11, float* __restrict__ x21) {
  int bg = blockIdx.x;
  int c = threadIdx.x & 31;
  float s1 = 0.f, s2 = 0.f;
  for (int j = 0; j < HB; ++j) {
    size_t o = ((size_t)bg * HB + j) * DD + c;
    s1 += ps1[o];
    s2 += ps2[o];
  }
  s1 *= 1.0f / (HH * WW);
  s2 *= 1.0f / (HH * WW);
  float m1 = wave_max32(s1);
  float e1 = __expf(s1 - m1);
  x11[bg * DD + c] = e1 * fast_rcp(wave_sum32(e1));
  float m2 = wave_max32(s2);
  float e2 = __expf(s2 - m2);
  x21[bg * DD + c] = e2 * fast_rcp(wave_sum32(e2));
}

// ------------------------------------------------------------------
// K5: recompute x1, gate = sigmoid(x11.x2 + x21.x1), out = gx * gate
// Wave32 per pixel, lane = channel.
// ------------------------------------------------------------------
__global__ void __launch_bounds__(256) k_final(
    const float* __restrict__ x, const float* __restrict__ hwg,
    const float* __restrict__ x2w, const float* __restrict__ x11,
    const float* __restrict__ x21, const float* __restrict__ gam,
    const float* __restrict__ bet, float* __restrict__ out) {
  int h = blockIdx.x, bg = blockIdx.y;
  int b = bg >> 3, gi = bg & 7;
  int wv = threadIdx.x >> 5, c = threadIdx.x & 31;
  const float* xb = x + ((size_t)b * HH + h) * WW * CC + (size_t)gi * DD;
  float* ob = out + ((size_t)b * HH + h) * WW * CC + (size_t)gi * DD;
  const float* hwb = hwg + (size_t)bg * (HH + WW) * DD;
  const float* x2row = x2w + ((size_t)bg * HH + h) * WW * DD;
  float xhv = hwb[h * DD + c];
  float a11 = x11[bg * DD + c], a21 = x21[bg * DD + c];
  float gm = gam[c], bt = bet[c];
  for (int p = wv; p < WW; p += 8) {
    float gxv = xb[(size_t)p * CC + c];
    float y = gxv * xhv * hwb[(HH + p) * DD + c];
    float mu = wave_sum32(y) * (1.0f / DD);
    float e2 = wave_sum32(y * y) * (1.0f / DD);
    float x1v = (y - mu) * fast_rsq(e2 - mu * mu + LN_EPSF) * gm + bt;
    float x2v = x2row[(size_t)p * DD + c];
    float w1s = wave_sum32(a11 * x2v);
    float w2s = wave_sum32(a21 * x1v);
    float gate = sigmoidf(w1s + w2s);
    ob[(size_t)p * CC + c] = gxv * gate;
  }
}

// ------------------------------------------------------------------
extern "C" void kernel_launch(void* const* d_in, const int* in_sizes, int n_in,
                              void* d_out, int out_size, void* d_ws, size_t ws_size,
                              hipStream_t stream) {
  (void)in_sizes; (void)n_in; (void)out_size; (void)ws_size;
  const float* x  = (const float*)d_in[0];
  const float* w1 = (const float*)d_in[1];
  const float* b1 = (const float*)d_in[2];
  const float* w3 = (const float*)d_in[3];
  const float* b3 = (const float*)d_in[4];
  const float* gm = (const float*)d_in[5];
  const float* bt = (const float*)d_in[6];
  float* out = (float*)d_out;
  float* ws  = (float*)d_ws;

  float* hwp = ws + OFF_HWP;
  float* hwg = ws + OFF_HWG;
  float* x11 = ws + OFF_X11;
  float* x21 = ws + OFF_X21;
  float* ps1 = ws + OFF_PS1;
  float* ps2 = ws + OFF_PS2;
  _Float16* wf3 = (_Float16*)(ws + OFF_WF3);
  _Float16* wf1 = (_Float16*)(ws + OFF_WF1);
  float* x2  = ws + OFF_X2;

  k_wprep<<<dim3(36), dim3(256), 0, stream>>>(w3, w1, wf3, wf1);
  k_pool<<<dim3(BG), dim3(256), 0, stream>>>(x, hwp);
  k_gate_gemm<<<dim3(BG * (HH + WW) / 16), dim3(32), 0, stream>>>(hwp, wf1, b1, hwg);
  k_main<<<dim3(HB, BG), dim3(96), 0, stream>>>(x, wf3, b3, gm, bt, hwg, x2, ps1, ps2);
  k_softmax<<<dim3(BG), dim3(32), 0, stream>>>(ps1, ps2, x11, x21);
  k_final<<<dim3(HH, BG), dim3(256), 0, stream>>>(x, hwg, x2, x11, x21, gm, bt, out);
}
